// MultiTaskDeep4Net_18597208391942
// MI455X (gfx1250) — matrix operand order repaired and checked by reading, not compiled, not run on hardware
//
#include <hip/hip_runtime.h>
#include <hip/hip_bf16.h>

typedef __attribute__((ext_vector_type(16))) _Float16 v16h;
typedef __attribute__((ext_vector_type(8)))  float    v8f;

#define EPS 1e-5f
#define BATCH 128

// ---------------------------------------------------------------------------
// CDNA5 async global->LDS copy support (toolchain-dependent builtin coverage)
// ---------------------------------------------------------------------------
#if defined(__has_builtin)
#  if __has_builtin(__builtin_amdgcn_global_load_async_to_lds_b32)
#    define HAVE_ASYNC_LDS 1
#  endif
#endif
#ifndef HAVE_ASYNC_LDS
#  define HAVE_ASYNC_LDS 0
#endif

#if defined(__has_builtin) && __has_builtin(__builtin_amdgcn_s_wait_asynccnt)
#  define WAIT_ASYNCCNT0() __builtin_amdgcn_s_wait_asynccnt(0)
#else
#  define WAIT_ASYNCCNT0() asm volatile("s_wait_asynccnt 0x0" ::: "memory")
#endif

typedef __attribute__((address_space(1))) int g_int;
typedef __attribute__((address_space(3))) int l_int;

// layer geometry: Tconv = Tin - 9 ; Tpool = Tconv/3
// L1: 64->25,  Tin=1000 Tconv=991 Tpool=330, K=640,  ksteps=20, mtiles=2
// L2: 25->50,  Tin=330  Tconv=321 Tpool=107, K=250,  ksteps=8,  mtiles=4
// L3: 50->100, Tin=107  Tconv=98  Tpool=32,  K=500,  ksteps=16, mtiles=7
// L4: 100->200,Tin=32   Tconv=23  Tpool=7,   K=1000, ksteps=32, mtiles=13

// ---------------------------------------------------------------------------
// Prep: fold conv_time into conv_spat -> effective conv weights (f16, packed
// in the exact 16x32 f16 A-fragment per-lane layout of V_WMMA_F32_16X16X32_F16)
// A-frag element for (lane, half h):  M = lane&15,
//   Klocal = h + 8*(h>>3) + 8*(lane>>4)   (ISA 7.12.2, 16-bit A 16x32)
// ---------------------------------------------------------------------------
__global__ void pack_weff_kernel(const float* __restrict__ w_time,   // (25,1,10)
                                 const float* __restrict__ w_spat,   // (25,25,64)
                                 _Float16* __restrict__ pack)        // [2][20][32][16]
{
    int idx = blockIdx.x * blockDim.x + threadIdx.x;
    const int total = 2 * 20 * 512;
    if (idx >= total) return;
    int h    = idx & 15;
    int lane = (idx >> 4) & 31;
    int ks   = (idx >> 9) % 20;
    int mt   = (idx >> 9) / 20;
    int m    = mt * 16 + (lane & 15);
    int kl   = h + ((h >> 3) << 3) + ((lane >> 4) << 3);
    int kidx = ks * 32 + kl;            // kidx = c*10 + tap
    float val = 0.f;
    if (m < 25 && kidx < 640) {
        int c   = kidx / 10;
        int tap = kidx - c * 10;
        float s = 0.f;
        for (int f = 0; f < 25; ++f)
            s += w_spat[(m * 25 + f) * 64 + c] * w_time[f * 10 + tap];
        val = s;
    }
    pack[idx] = (_Float16)val;
}

__global__ void compute_beff_kernel(const float* __restrict__ w_spat,
                                    const float* __restrict__ b_time,
                                    float* __restrict__ beff)        // (25,)
{
    int g = threadIdx.x;
    if (g >= 25) return;
    float s = 0.f;
    for (int f = 0; f < 25; ++f) {
        float acc = 0.f;
        for (int c = 0; c < 64; ++c) acc += w_spat[(g * 25 + f) * 64 + c];
        s += acc * b_time[f];
    }
    beff[g] = s;
}

// Generic conv-weight packer: w (Cout, Cin, 10) fp32 -> A-fragment f16
__global__ void pack_w_kernel(const float* __restrict__ w, _Float16* __restrict__ pack,
                              int Cout, int Cin, int ksteps, int mtiles)
{
    int idx = blockIdx.x * blockDim.x + threadIdx.x;
    int total = mtiles * ksteps * 512;
    if (idx >= total) return;
    int h    = idx & 15;
    int lane = (idx >> 4) & 31;
    int ks   = (idx >> 9) % ksteps;
    int mt   = (idx >> 9) / ksteps;
    int m    = mt * 16 + (lane & 15);
    int kl   = h + ((h >> 3) << 3) + ((lane >> 4) << 3);
    int kidx = ks * 32 + kl;
    int K    = Cin * 10;
    float val = 0.f;
    if (m < Cout && kidx < K) {
        int c   = kidx / 10;
        int tap = kidx - c * 10;
        val = w[(m * Cin + c) * 10 + tap];
    }
    pack[idx] = (_Float16)val;
}

// ---------------------------------------------------------------------------
// Fused conv1d (valid, taps=10) + optional bias + BN(eval) + ELU, via WMMA.
// Fully specialized per layer. Grid: (B, ceil(TCONV/16)); block: 128 = 4 waves.
// Staging uses GLOBAL_LOAD_ASYNC_TO_LDS_B32 (ASYNCcnt) when available.
// B-fragment per-lane layout (K x N = 32 x 16): N = lane&15,
//   Klocal = h + 8*(h>>3) + 8*(lane>>4)   (matches A-fragment K order)
// ---------------------------------------------------------------------------
template <int CIN, int COUT, int TIN, int TCONV, int KSTEPS, int MTILES, bool HAS_BIAS>
__global__ __launch_bounds__(128)
void conv_wmma_bn_elu_kernel(const float* __restrict__ in,        // (B, CIN, TIN)
                             const _Float16* __restrict__ wpack,  // [MTILES][KSTEPS][32][16]
                             const float* __restrict__ cbias,     // (COUT,) if HAS_BIAS
                             const float* __restrict__ gn,        // BN gamma
                             const float* __restrict__ bt,        // BN beta
                             const float* __restrict__ mn,        // BN mean
                             const float* __restrict__ vr,        // BN var
                             float* __restrict__ out)             // (B, COUT, TCONV)
{
    constexpr int TW  = 25;              // 16 outputs + 10 taps - 1
    constexpr int K   = CIN * 10;
    __shared__ float sIn[CIN * TW];

    const int b    = blockIdx.x;
    const int t0   = blockIdx.y * 16;
    const int tid  = threadIdx.x;
    const int lane = tid & 31;
    const int wid  = tid >> 5;
    const float* inb = in + (size_t)b * CIN * TIN;

#if HAVE_ASYNC_LDS
    for (int i = tid; i < CIN * TW; i += 128) {
        int c = i / TW;
        int j = i - c * TW;
        int t = t0 + j;
        if (t < TIN) {
            __builtin_amdgcn_global_load_async_to_lds_b32(
                (g_int*)(inb + c * TIN + t),
                (l_int*)&sIn[i],
                /*offset=*/0, /*cpol=*/0);
        } else {
            sIn[i] = 0.f;                // tail zero-fill via DS store
        }
        int tn = t + 16;                 // prefetch next time tile
        if (tn < TIN) __builtin_prefetch(inb + c * TIN + tn, 0, 2);
    }
    WAIT_ASYNCCNT0();                    // async LDS writes visible pre-barrier
#else
    for (int i = tid; i < CIN * TW; i += 128) {
        int c = i / TW;
        int j = i - c * TW;
        int t = t0 + j;
        sIn[i] = (t < TIN) ? inb[c * TIN + t] : 0.f;
        int tn = t + 16;
        if (tn < TIN) __builtin_prefetch(inb + c * TIN + tn, 0, 2);
    }
#endif
    __syncthreads();

    const int n     = lane & 15;         // output column (time)
    const int khalf = (lane >> 4) << 3;  // B-frag K half select (8*(lane>>4))

    v8f acc[4];
#pragma unroll
    for (int i = 0; i < 4; ++i) acc[i] = (v8f)0.0f;

    for (int ks = 0; ks < KSTEPS; ++ks) {
        // ---- build B fragment from LDS ----
        v16h bvf;
        int kbase = ks * 32 + khalf;
#pragma unroll
        for (int h = 0; h < 16; ++h) {
            int kidx = kbase + h + ((h >> 3) << 3);
            int kc   = (kidx < K) ? kidx : 0;
            int c    = kc / 10;          // constant division -> mul/shift
            int tp   = kc - c * 10;
            float f  = (kidx < K) ? sIn[c * TW + n + tp] : 0.f;
            bvf[h] = (_Float16)f;
        }
        // ---- up to 4 M-tiles per wave share this B fragment ----
#pragma unroll
        for (int i = 0; i < 4; ++i) {
            int mt = wid + 4 * i;
            if (mt < MTILES) {
                v16h av = *(const v16h*)(wpack + (((size_t)mt * KSTEPS + ks) * 32 + lane) * 16);
                acc[i] = __builtin_amdgcn_wmma_f32_16x16x32_f16(
                    /*neg_a=*/false, av, /*neg_b=*/false, bvf,
                    /*c_mod=*/(short)0, acc[i], /*reuse_a=*/false, /*reuse_b=*/false);
            }
        }
    }

    // ---- epilogue: bias + BN(eval) + ELU, scatter per C/D layout ----
    const int rowbase = (lane >> 4) * 8; // lanes 16-31 hold rows M=8..15
    const int t = t0 + n;
#pragma unroll
    for (int i = 0; i < 4; ++i) {
        int mt = wid + 4 * i;
        if (mt < MTILES) {
#pragma unroll
            for (int r = 0; r < 8; ++r) {
                int m = mt * 16 + rowbase + r;
                if (m < COUT && t < TCONV) {
                    float val = acc[i][r];
                    if (HAS_BIAS) val += cbias[m];
                    float inv = __frsqrt_rn(vr[m] + EPS);
                    val = (val - mn[m]) * (gn[m] * inv) + bt[m];
                    val = (val > 0.f) ? val : (__expf(val) - 1.f);
                    out[((size_t)b * COUT + m) * TCONV + t] = val;
                }
            }
        }
    }
}

// ---------------------------------------------------------------------------
// MaxPool k=3 s=3 along time: (BC, TC) -> (BC, TP); geometry compile-time.
// ---------------------------------------------------------------------------
template <int TC, int TP, int TOTAL>
__global__ void pool3_kernel(const float* __restrict__ in, float* __restrict__ out)
{
    int idx = blockIdx.x * blockDim.x + threadIdx.x;
    if (idx >= TOTAL) return;
    int p  = idx % TP;                   // constant division -> mul/shift
    int bc = idx / TP;
    const float* r = in + (size_t)bc * TC + p * 3;
    out[idx] = fmaxf(fmaxf(r[0], r[1]), r[2]);
}

// ---------------------------------------------------------------------------
// Classifier conv (time->1) + per-cluster expert head. One block per sample.
// ---------------------------------------------------------------------------
__global__ __launch_bounds__(256)
void cls_head_kernel(const float* __restrict__ z4p,    // (B, 200, 7)
                     const float* __restrict__ w_cls,  // (4, 200, 7)
                     const float* __restrict__ b_cls,  // (4,)
                     const int*   __restrict__ cids,   // (B,)
                     const float* __restrict__ hW1,    // (8, 512, 4)
                     const float* __restrict__ hb1,    // (8, 512)
                     const float* __restrict__ hW2,    // (8, 4, 512)
                     const float* __restrict__ hb2,    // (8, 4)
                     const float* __restrict__ hg, const float* __restrict__ hbe,
                     const float* __restrict__ hm, const float* __restrict__ hv,
                     float* __restrict__ out)          // (B, 4)
{
    __shared__ float sfeat[4];
    __shared__ float sh1[512];

    const int b    = blockIdx.x;
    const int tid  = threadIdx.x;
    const int lane = tid & 31;
    const int wid  = tid >> 5;
    const float* zb = z4p + (size_t)b * 1400;

    // feats[o] = <w_cls[o,:,:], z4p[b,:,:]> + b_cls[o]
    if (wid < 4) {
        float s = 0.f;
        for (int i = lane; i < 1400; i += 32)
            s += w_cls[wid * 1400 + i] * zb[i];
        for (int off = 16; off > 0; off >>= 1)
            s += __shfl_down(s, off, 32);
        if (lane == 0) sfeat[wid] = s + b_cls[wid];
    }
    __syncthreads();

    const int cid = cids[b];
    const float f0 = sfeat[0], f1 = sfeat[1], f2 = sfeat[2], f3 = sfeat[3];

    // h1 = relu(BN(hW1[cid] @ feats + hb1[cid]))
    for (int h = tid; h < 512; h += 256) {
        const int    hi = cid * 512 + h;
        const float* w  = hW1 + (size_t)hi * 4;
        float val = w[0] * f0 + w[1] * f1 + w[2] * f2 + w[3] * f3 + hb1[hi];
        float inv = __frsqrt_rn(hv[hi] + EPS);
        val = (val - hm[hi]) * (hg[hi] * inv) + hbe[hi];
        sh1[h] = (val > 0.f) ? val : 0.f;
    }
    __syncthreads();

    // out = hW2[cid] @ h1 + hb2[cid]
    if (wid < 4) {
        float s = 0.f;
        for (int h = lane; h < 512; h += 32)
            s += hW2[((size_t)cid * 4 + wid) * 512 + h] * sh1[h];
        for (int off = 16; off > 0; off >>= 1)
            s += __shfl_down(s, off, 32);
        if (lane == 0) out[b * 4 + wid] = s + hb2[cid * 4 + wid];
    }
}

// ---------------------------------------------------------------------------
// Launch
// ---------------------------------------------------------------------------
extern "C" void kernel_launch(void* const* d_in, const int* in_sizes, int n_in,
                              void* d_out, int out_size, void* d_ws, size_t ws_size,
                              hipStream_t stream)
{
    const float* x       = (const float*)d_in[0];
    const int*   cids    = (const int*)  d_in[1];
    const float* w_time  = (const float*)d_in[2];
    const float* b_time  = (const float*)d_in[3];
    const float* w_spat  = (const float*)d_in[4];
    const float* w2      = (const float*)d_in[5];
    const float* w3      = (const float*)d_in[6];
    const float* w4      = (const float*)d_in[7];
    const float* w_cls   = (const float*)d_in[8];
    const float* b_cls   = (const float*)d_in[9];
    const float* hW1     = (const float*)d_in[10];
    const float* hb1     = (const float*)d_in[11];
    const float* hW2     = (const float*)d_in[12];
    const float* hb2     = (const float*)d_in[13];
    const float* hg      = (const float*)d_in[14];
    const float* hbe     = (const float*)d_in[15];
    const float* hm      = (const float*)d_in[16];
    const float* hv      = (const float*)d_in[17];
    const float* g1  = (const float*)d_in[18], *be1 = (const float*)d_in[19];
    const float* m1  = (const float*)d_in[20], *v1  = (const float*)d_in[21];
    const float* g2  = (const float*)d_in[22], *be2 = (const float*)d_in[23];
    const float* m2  = (const float*)d_in[24], *v2  = (const float*)d_in[25];
    const float* g3  = (const float*)d_in[26], *be3 = (const float*)d_in[27];
    const float* m3  = (const float*)d_in[28], *v3  = (const float*)d_in[29];
    const float* g4  = (const float*)d_in[30], *be4 = (const float*)d_in[31];
    const float* m4  = (const float*)d_in[32], *v4  = (const float*)d_in[33];
    float* out = (float*)d_out;
    (void)n_in; (void)in_sizes; (void)out_size; (void)ws_size;

    // ---- workspace carve-out (256B aligned) ----
    char* ws = (char*)d_ws;
    size_t off = 0;
    auto carve = [&](size_t bytes) -> void* {
        void* p = (void*)(ws + off);
        off += (bytes + 255) & ~(size_t)255;
        return p;
    };
    _Float16* wp1  = (_Float16*)carve((size_t)2  * 20 * 512 * 2);
    _Float16* wp2  = (_Float16*)carve((size_t)4  * 8  * 512 * 2);
    _Float16* wp3  = (_Float16*)carve((size_t)7  * 16 * 512 * 2);
    _Float16* wp4  = (_Float16*)carve((size_t)13 * 32 * 512 * 2);
    float*    beff = (float*)   carve(25 * 4);
    float*    z1   = (float*)   carve((size_t)BATCH * 25  * 991 * 4);
    float*    z1p  = (float*)   carve((size_t)BATCH * 25  * 330 * 4);
    float*    z2   = (float*)   carve((size_t)BATCH * 50  * 321 * 4);
    float*    z2p  = (float*)   carve((size_t)BATCH * 50  * 107 * 4);
    float*    z3   = (float*)   carve((size_t)BATCH * 100 * 98  * 4);
    float*    z3p  = (float*)   carve((size_t)BATCH * 100 * 32  * 4);
    float*    z4   = (float*)   carve((size_t)BATCH * 200 * 23  * 4);
    float*    z4p  = (float*)   carve((size_t)BATCH * 200 * 7   * 4);

    // ---- weight prep ----
    compute_beff_kernel<<<1, 32, 0, stream>>>(w_spat, b_time, beff);
    pack_weff_kernel<<<(2 * 20 * 512 + 255) / 256, 256, 0, stream>>>(w_time, w_spat, wp1);
    pack_w_kernel<<<(4  * 8  * 512 + 255) / 256, 256, 0, stream>>>(w2, wp2,  50,  25,  8,  4);
    pack_w_kernel<<<(7  * 16 * 512 + 255) / 256, 256, 0, stream>>>(w3, wp3, 100,  50, 16,  7);
    pack_w_kernel<<<(13 * 32 * 512 + 255) / 256, 256, 0, stream>>>(w4, wp4, 200, 100, 32, 13);

    // ---- layer 1: fused (conv_time ∘ conv_spat) + BN + ELU, then pool ----
    conv_wmma_bn_elu_kernel<64, 25, 1000, 991, 20, 2, true>
        <<<dim3(BATCH, (991 + 15) / 16), 128, 0, stream>>>(
            x, wp1, beff, g1, be1, m1, v1, z1);
    pool3_kernel<991, 330, BATCH * 25 * 330>
        <<<(BATCH * 25 * 330 + 255) / 256, 256, 0, stream>>>(z1, z1p);

    // ---- layer 2 ----
    conv_wmma_bn_elu_kernel<25, 50, 330, 321, 8, 4, false>
        <<<dim3(BATCH, (321 + 15) / 16), 128, 0, stream>>>(
            z1p, wp2, nullptr, g2, be2, m2, v2, z2);
    pool3_kernel<321, 107, BATCH * 50 * 107>
        <<<(BATCH * 50 * 107 + 255) / 256, 256, 0, stream>>>(z2, z2p);

    // ---- layer 3 ----
    conv_wmma_bn_elu_kernel<50, 100, 107, 98, 16, 7, false>
        <<<dim3(BATCH, (98 + 15) / 16), 128, 0, stream>>>(
            z2p, wp3, nullptr, g3, be3, m3, v3, z3);
    pool3_kernel<98, 32, BATCH * 100 * 32>
        <<<(BATCH * 100 * 32 + 255) / 256, 256, 0, stream>>>(z3, z3p);

    // ---- layer 4 ----
    conv_wmma_bn_elu_kernel<100, 200, 32, 23, 32, 13, false>
        <<<dim3(BATCH, (23 + 15) / 16), 128, 0, stream>>>(
            z3p, wp4, nullptr, g4, be4, m4, v4, z4);
    pool3_kernel<23, 7, BATCH * 200 * 7>
        <<<(BATCH * 200 * 7 + 255) / 256, 256, 0, stream>>>(z4, z4p);

    // ---- classifier + per-cluster expert head ----
    cls_head_kernel<<<BATCH, 256, 0, stream>>>(z4p, w_cls, b_cls, cids,
                                               hW1, hb1, hW2, hb2, hg, hbe, hm, hv, out);
}
